// MaskedMultiHeadAttentionBlock_84688165143272
// MI455X (gfx1250) — matrix operand order repaired and checked by reading, not compiled, not run on hardware
//
#include <hip/hip_runtime.h>
#include <stdint.h>

#define BB  2
#define SS  2048
#define DDm 1024
#define HHm 16
#define DKK 64

typedef __attribute__((ext_vector_type(16))) __bf16 v16bf;
typedef __attribute__((ext_vector_type(8)))  __bf16 v8bf;
typedef __attribute__((ext_vector_type(8)))  float  v8f;

static __device__ __forceinline__ __bf16 f2bf(float f) {
  unsigned u = __builtin_bit_cast(unsigned, f);
  unsigned r = (u + 0x7FFFu + ((u >> 16) & 1u)) >> 16;
  unsigned short h = (unsigned short)r;
  return __builtin_bit_cast(__bf16, h);
}

static __device__ __forceinline__ v16bf mk16(v8bf lo, v8bf hi) {
  v16bf r;
#pragma unroll
  for (int i = 0; i < 8; ++i) { r[i] = lo[i]; r[i + 8] = hi[i]; }
  return r;
}

static __device__ __forceinline__ v8f vzero8() {
  v8f z;
#pragma unroll
  for (int i = 0; i < 8; ++i) z[i] = 0.f;
  return z;
}

// ---- CDNA5 async global->LDS copy (ASYNCcnt), CDNA5 ISA ----
// VDST vgpr = per-lane LDS byte offset; VADDR = 64-bit global byte address.
static __device__ __forceinline__ void async_cp16(uint32_t lds_off, const void* g) {
  uint64_t ga = (uint64_t)(uintptr_t)g;
  asm volatile("global_load_async_to_lds_b128 %0, %1, off"
               :: "v"(lds_off), "v"(ga) : "memory");
}
static __device__ __forceinline__ void wait_async0() {
  asm volatile("s_wait_asynccnt 0x0" ::: "memory");
}

__global__ void cvt_f32_bf16(const float* __restrict__ src, __bf16* __restrict__ dst, int n) {
  int i = blockIdx.x * blockDim.x + threadIdx.x;
  if (i < n) dst[i] = f2bf(src[i]);
}

// Y = X(MxK row-major) * W^T, W stored [N,K] row-major. M=4096, N=K=1024.
// Workgroup (8 waves) computes a 128x128 tile; A/B k-slices double-buffered in
// LDS via async copies; each wave computes 64x32 (4x2 WMMA tiles).
// mode 0/1: bf16 out, head-major [B,H,S,DK] (mode 0 applies `scale`)
// mode 2  : bf16 out, transposed  [B,H,DK,S]
// mode 3  : f32 out, [B,S,D]
__global__ void proj_gemm(const __bf16* __restrict__ X, const __bf16* __restrict__ W,
                          __bf16* __restrict__ Yh, float* __restrict__ Yf,
                          int mode, float scale) {
  __shared__ __bf16 Asm[2][128 * 32];   // [buf][row*32 + k], 8KB each
  __shared__ __bf16 Bsm[2][128 * 32];

  const int tid   = threadIdx.x;
  const int lane  = tid & 31;
  const int wib   = tid >> 5;           // 0..7
  const int mwave = (wib & 1) * 64;     // wave's M offset inside 128
  const int nwave = (wib >> 1) * 32;    // wave's N offset inside 128
  const int hl    = lane >> 4;
  const int l15   = lane & 15;

  const int bm = blockIdx.x >> 3;       // M/128 = 32
  const int bn = blockIdx.x & 7;        // N/128 = 8
  const int m0 = bm * 128, n0 = bn * 128;

  const uint32_t aBase0 = (uint32_t)(uintptr_t)&Asm[0][0];
  const uint32_t aBase1 = (uint32_t)(uintptr_t)&Asm[1][0];
  const uint32_t bBase0 = (uint32_t)(uintptr_t)&Bsm[0][0];
  const uint32_t bBase1 = (uint32_t)(uintptr_t)&Bsm[1][0];

  auto issue_slice = [&](int k0, int buf) {
    uint32_t ab = buf ? aBase1 : aBase0;
    uint32_t bb = buf ? bBase1 : bBase0;
#pragma unroll
    for (int cc = 0; cc < 2; ++cc) {
      int c   = tid + cc * 256;
      int row = c >> 2;
      int cq  = c & 3;                       // 16B quarter -> 8 bf16 elements
      async_cp16(ab + (uint32_t)(row * 64 + cq * 16),
                 X + (size_t)(m0 + row) * DDm + k0 + cq * 8);
      async_cp16(bb + (uint32_t)(row * 64 + cq * 16),
                 W + (size_t)(n0 + row) * DDm + k0 + cq * 8);
    }
  };

  v8f acc[4][2];
#pragma unroll
  for (int mt = 0; mt < 4; ++mt)
#pragma unroll
    for (int nt = 0; nt < 2; ++nt) acc[mt][nt] = vzero8();

  issue_slice(0, 0);
  wait_async0();
  __syncthreads();

  const int KSTEPS = DDm / 32;            // 32
  for (int ks = 0; ks < KSTEPS; ++ks) {
    const int cur = ks & 1;
    if (ks + 1 < KSTEPS) issue_slice((ks + 1) * 32, cur ^ 1);

    v16bf afrag[4], bfrag[2];
#pragma unroll
    for (int mt = 0; mt < 4; ++mt) {
      const __bf16* ar = &Asm[cur][(mwave + mt * 16 + l15) * 32];
      afrag[mt] = mk16(*(const v8bf*)(ar + hl * 8), *(const v8bf*)(ar + 16 + hl * 8));
    }
#pragma unroll
    for (int nt = 0; nt < 2; ++nt) {
      const __bf16* br = &Bsm[cur][(nwave + nt * 16 + l15) * 32];
      bfrag[nt] = mk16(*(const v8bf*)(br + hl * 8), *(const v8bf*)(br + 16 + hl * 8));
    }
#pragma unroll
    for (int mt = 0; mt < 4; ++mt)
#pragma unroll
      for (int nt = 0; nt < 2; ++nt)
        acc[mt][nt] = __builtin_amdgcn_wmma_f32_16x16x32_bf16(
            false, afrag[mt], false, bfrag[nt], (short)0, acc[mt][nt], false, false);

    wait_async0();        // prefetched slice resident in LDS
    __syncthreads();      // all waves done reading `cur` before it is rewritten
  }

#pragma unroll
  for (int mt = 0; mt < 4; ++mt) {
#pragma unroll
    for (int nt = 0; nt < 2; ++nt) {
#pragma unroll
      for (int r = 0; r < 8; ++r) {
        int row = m0 + mwave + mt * 16 + r + 8 * hl;   // C layout: VGPR r -> row r+8*half
        int col = n0 + nwave + nt * 16 + l15;
        float val = acc[mt][nt][r] * scale;
        if (mode == 3) {
          Yf[(size_t)row * DDm + col] = val;
        } else {
          int bidx = row >> 11;              // / S
          int sidx = row & (SS - 1);
          int hidx = col >> 6;               // / DK
          int dk   = col & (DKK - 1);
          size_t idx;
          if (mode == 2) idx = (((size_t)(bidx * HHm + hidx)) * DKK + dk) * SS + sidx;
          else           idx = (((size_t)(bidx * HHm + hidx)) * SS + sidx) * DKK + dk;
          Yh[idx] = f2bf(val);
        }
      }
    }
  }
}

// Flash attention, causal. Workgroup (8 waves) covers 8 consecutive 16-row
// query tiles of one (b,h). K/V 32-key tiles are staged once per workgroup in
// LDS (double-buffered async copies) and shared by all 8 waves; waves past
// their causal limit skip compute but join the async-wait + barrier.
// Qh,Kh: [B,H,S,DK] bf16 (Q pre-scaled by 1/sqrt(DK)); Vt: [B,H,DK,S] bf16.
// An: [B,S,D] bf16 output (pre output-projection).
__global__ void attn_fwd(const __bf16* __restrict__ Qh, const __bf16* __restrict__ Kh,
                         const __bf16* __restrict__ Vt, __bf16* __restrict__ An) {
  __shared__ __bf16 Ksm[2][32 * 64];       // [buf][j*64 + dk], 4KB each
  __shared__ __bf16 Vsm[2][64 * 32];       // [buf][dk*32 + j], 4KB each
  __shared__ __bf16 plds[8 * 16 * 32];     // per-wave 16x32 bf16 P staging

  const int tid  = threadIdx.x;
  const int lane = tid & 31;
  const int wib  = tid >> 5;               // 0..7
  const int bh   = blockIdx.x >> 4;        // 0..B*H-1
  const int qg   = blockIdx.x & 15;        // q-group (8 q-tiles each)
  const int bidx = bh >> 4, hidx = bh & 15;
  const int q0   = (qg * 8 + wib) * 16;    // this wave's query-tile start
  const int hl   = lane >> 4;
  const int l15  = lane & 15;

  const __bf16* Qb = Qh + (size_t)bh * SS * DKK;
  const __bf16* Kb = Kh + (size_t)bh * SS * DKK;
  const __bf16* Vb = Vt + (size_t)bh * DKK * SS;
  __bf16* pl = &plds[wib * 16 * 32];

  const uint32_t kB0 = (uint32_t)(uintptr_t)&Ksm[0][0];
  const uint32_t kB1 = (uint32_t)(uintptr_t)&Ksm[1][0];
  const uint32_t vB0 = (uint32_t)(uintptr_t)&Vsm[0][0];
  const uint32_t vB1 = (uint32_t)(uintptr_t)&Vsm[1][0];

  // One 16B chunk of K and one of V per thread per 32-key block.
  auto issue_kv = [&](int k0, int buf) {
    uint32_t kb = buf ? kB1 : kB0;
    uint32_t vb = buf ? vB1 : vB0;
    {
      int row = tid >> 3, cq = tid & 7;    // K: 32 rows x 128B
      async_cp16(kb + (uint32_t)(row * 128 + cq * 16),
                 Kb + (size_t)(k0 + row) * DKK + cq * 8);
    }
    {
      int row = tid >> 2, cq = tid & 3;    // V: 64 rows x 64B
      async_cp16(vb + (uint32_t)(row * 64 + cq * 16),
                 Vb + (size_t)row * SS + k0 + cq * 8);
    }
  };

  // Q A-fragments for both K-chunks (DK=64 -> 2 chunks of 32)
  v16bf aq[2];
#pragma unroll
  for (int kc = 0; kc < 2; ++kc) {
    const __bf16* qr = Qb + (size_t)(q0 + l15) * DKK + kc * 32;
    aq[kc] = mk16(*(const v8bf*)(qr + hl * 8), *(const v8bf*)(qr + 16 + hl * 8));
  }

  v8f o[4];
#pragma unroll
  for (int t = 0; t < 4; ++t) o[t] = vzero8();
  float mstat[8], lstat[8];
#pragma unroll
  for (int r = 0; r < 8; ++r) { mstat[r] = -3.0e38f; lstat[r] = 0.f; }

  issue_kv(0, 0);
  wait_async0();
  __syncthreads();

  const int nblocks = qg * 4 + 4;          // covers keys 0 .. qg*128+127
  for (int kbi = 0; kbi < nblocks; ++kbi) {
    const int k0  = kbi * 32;
    const int cur = kbi & 1;
    if (kbi + 1 < nblocks) issue_kv(k0 + 32, cur ^ 1);

    if (k0 <= q0 + 15) {                   // wave-uniform causal skip
      // ---- scores S = Q K^T from LDS K tile ----
      v8f s[2];
#pragma unroll
      for (int jt = 0; jt < 2; ++jt) {
        s[jt] = vzero8();
        const __bf16* kr = &Ksm[cur][(jt * 16 + l15) * 64 + hl * 8];
        v16bf b0 = mk16(*(const v8bf*)(kr), *(const v8bf*)(kr + 16));
        v16bf b1 = mk16(*(const v8bf*)(kr + 32), *(const v8bf*)(kr + 48));
        s[jt] = __builtin_amdgcn_wmma_f32_16x16x32_bf16(false, aq[0], false, b0,
                                                        (short)0, s[jt], false, false);
        s[jt] = __builtin_amdgcn_wmma_f32_16x16x32_bf16(false, aq[1], false, b1,
                                                        (short)0, s[jt], false, false);
      }

      // ---- causal mask + online softmax (row stats replicated per half-wave) ----
      float rowscale[8];
      const int jbase = k0 + l15;
#pragma unroll
      for (int r = 0; r < 8; ++r) {
        int row = q0 + r + 8 * hl;
        float s0 = (jbase      <= row) ? s[0][r] : -1.0e30f;
        float s1 = (jbase + 16 <= row) ? s[1][r] : -1.0e30f;
        float rm = fmaxf(s0, s1);
#pragma unroll
        for (int m = 1; m < 16; m <<= 1) rm = fmaxf(rm, __shfl_xor(rm, m, 32));
        float mnew  = fmaxf(mstat[r], rm);
        float alpha = __expf(mstat[r] - mnew);
        float p0 = __expf(s0 - mnew);
        float p1 = __expf(s1 - mnew);
        float rs = p0 + p1;
#pragma unroll
        for (int m = 1; m < 16; m <<= 1) rs += __shfl_xor(rs, m, 32);
        lstat[r] = lstat[r] * alpha + rs;
        mstat[r] = mnew;
        rowscale[r] = alpha;
        s[0][r] = p0; s[1][r] = p1;
      }
#pragma unroll
      for (int t = 0; t < 4; ++t)
#pragma unroll
        for (int r = 0; r < 8; ++r) o[t][r] *= rowscale[r];

      // ---- P: C-layout regs -> per-wave LDS -> A-layout fragment ----
#pragma unroll
      for (int jt = 0; jt < 2; ++jt)
#pragma unroll
        for (int r = 0; r < 8; ++r)
          pl[(r + 8 * hl) * 32 + jt * 16 + l15] = f2bf(s[jt][r]);
      __asm__ volatile("s_wait_dscnt 0" ::: "memory");
      const __bf16* pr = pl + l15 * 32 + hl * 8;
      v16bf pa = mk16(*(const v8bf*)(pr), *(const v8bf*)(pr + 16));

      // ---- O += P * V from LDS V tile (pre-transposed) ----
#pragma unroll
      for (int t = 0; t < 4; ++t) {
        const __bf16* vr = &Vsm[cur][(t * 16 + l15) * 32 + hl * 8];
        v16bf bv = mk16(*(const v8bf*)(vr), *(const v8bf*)(vr + 16));
        o[t] = __builtin_amdgcn_wmma_f32_16x16x32_bf16(false, pa, false, bv,
                                                       (short)0, o[t], false, false);
      }
    }

    wait_async0();
    __syncthreads();
  }

  // ---- normalize and store attn output [B,S,D] bf16 ----
#pragma unroll
  for (int t = 0; t < 4; ++t)
#pragma unroll
    for (int r = 0; r < 8; ++r) {
      int row = q0 + r + 8 * hl;
      int col = hidx * 64 + t * 16 + l15;
      float val = o[t][r] / lstat[r];
      An[((size_t)(bidx * SS + row)) * DDm + col] = f2bf(val);
    }
}

extern "C" void kernel_launch(void* const* d_in, const int* in_sizes, int n_in,
                              void* d_out, int out_size, void* d_ws, size_t ws_size,
                              hipStream_t stream) {
  const float* q  = (const float*)d_in[0];
  const float* k  = (const float*)d_in[1];
  const float* v  = (const float*)d_in[2];
  // d_in[3] is the causal mask; causality is computed analytically in attn_fwd.
  const float* Wq = (const float*)d_in[4];
  const float* Wk = (const float*)d_in[5];
  const float* Wv = (const float*)d_in[6];
  const float* Wo = (const float*)d_in[7];
  float* out = (float*)d_out;

  char* ws = (char*)d_ws;
  size_t off = 0;
  auto carve = [&](size_t elems) -> __bf16* {
    __bf16* p = (__bf16*)(ws + off);
    off += ((elems * sizeof(__bf16)) + 255) & ~(size_t)255;
    return p;
  };
  const size_t BSD  = (size_t)BB * SS * DDm;   // 4,194,304
  const size_t DDsz = (size_t)DDm * DDm;       // 1,048,576

  __bf16* qb  = carve(BSD);
  __bf16* kb  = carve(BSD);
  __bf16* vb  = carve(BSD);
  __bf16* wqb = carve(DDsz);
  __bf16* wkb = carve(DDsz);
  __bf16* wvb = carve(DDsz);
  __bf16* wob = carve(DDsz);
  __bf16* Qhd = carve(BSD);   // [B,H,S,DK], pre-scaled by 1/sqrt(DK)
  __bf16* Khd = carve(BSD);   // [B,H,S,DK]
  __bf16* Vtr = carve(BSD);   // [B,H,DK,S]
  __bf16* Ann = carve(BSD);   // [B,S,D]

  const int thr = 256;
  cvt_f32_bf16<<<(int)((BSD  + thr - 1) / thr), thr, 0, stream>>>(q,  qb,  (int)BSD);
  cvt_f32_bf16<<<(int)((BSD  + thr - 1) / thr), thr, 0, stream>>>(k,  kb,  (int)BSD);
  cvt_f32_bf16<<<(int)((BSD  + thr - 1) / thr), thr, 0, stream>>>(v,  vb,  (int)BSD);
  cvt_f32_bf16<<<(int)((DDsz + thr - 1) / thr), thr, 0, stream>>>(Wq, wqb, (int)DDsz);
  cvt_f32_bf16<<<(int)((DDsz + thr - 1) / thr), thr, 0, stream>>>(Wk, wkb, (int)DDsz);
  cvt_f32_bf16<<<(int)((DDsz + thr - 1) / thr), thr, 0, stream>>>(Wv, wvb, (int)DDsz);
  cvt_f32_bf16<<<(int)((DDsz + thr - 1) / thr), thr, 0, stream>>>(Wo, wob, (int)DDsz);

  // 256 workgroups x 256 threads: each computes a 128x128 output tile.
  proj_gemm<<<256, 256, 0, stream>>>(qb,  wqb, Qhd, nullptr, 0, 0.125f); // 1/sqrt(64)
  proj_gemm<<<256, 256, 0, stream>>>(kb,  wkb, Khd, nullptr, 1, 1.0f);
  proj_gemm<<<256, 256, 0, stream>>>(vb,  wvb, Vtr, nullptr, 2, 1.0f);

  // 512 blocks: each covers 8 consecutive q-tiles of one (b,h), LDS-shared K/V.
  attn_fwd<<<512, 256, 0, stream>>>(Qhd, Khd, Vtr, Ann);

  proj_gemm<<<256, 256, 0, stream>>>(Ann, wob, nullptr, out, 3, 1.0f);
}
